// GaussianUpsampling_14078902796991
// MI455X (gfx1250) — matrix operand order repaired and checked by reading, not compiled, not run on hardware
//
#include <hip/hip_runtime.h>
#include <hip/hip_bf16.h>

// ---------------- problem constants (from reference) ----------------
#define BATCH   16
#define T_TEXT  1024
#define T_FEAT  4096
#define CCH     512
#define DELTA   0.1f
#define K_CHUNK 32
#define P_STRIDE 40          // halves; 80B row stride -> conflict-free 16B LDS reads
#define RADIUS  16.0f        // exp(-0.1*16^2) ~ 8e-12: negligible softmax mass
#define WIN     40.0f        // expf(-0.1*40^2) == 0.0f exactly: bit-exact window
#define Y_ELEMS ((size_t)BATCH * T_FEAT * CCH)   // 33,554,432

typedef __attribute__((ext_vector_type(16))) _Float16 v16h;
typedef __attribute__((ext_vector_type(8)))  _Float16 v8h;
typedef __attribute__((ext_vector_type(8)))  float    v8f;

// ---------------- phase 1: per-row scan -> centers, lengths ----------------
__global__ void __launch_bounds__(256)
gu_scan_kernel(const int* __restrict__ dur, float* __restrict__ c,
               int* __restrict__ lens, int* __restrict__ lensOut) {
    const int b   = blockIdx.x;
    const int tid = threadIdx.x;          // 256 threads, 4 elements each
    __shared__ int part[256];

    int4 dv = ((const int4*)(dur + b * T_TEXT))[tid];
    int s0 = dv.x;
    int s1 = s0 + dv.y;
    int s2 = s1 + dv.z;
    int s3 = s2 + dv.w;
    part[tid] = s3;
    __syncthreads();

    int val = s3;                          // inclusive scan over 256 chunk sums
    for (int off = 1; off < 256; off <<= 1) {
        int add = (tid >= off) ? part[tid - off] : 0;
        __syncthreads();
        val += add;
        part[tid] = val;
        __syncthreads();
    }
    const int base = val - s3;             // exclusive prefix of this chunk
    float* cb = c + b * T_TEXT;
    const int j0 = tid * 4;
    cb[j0 + 0] = (float)(base + s0) - 0.5f * (float)dv.x;
    cb[j0 + 1] = (float)(base + s1) - 0.5f * (float)dv.y;
    cb[j0 + 2] = (float)(base + s2) - 0.5f * (float)dv.z;
    cb[j0 + 3] = (float)(base + s3) - 0.5f * (float)dv.w;

    if (tid == 0) {
        int total = part[255];
        lens[b]    = total;
        lensOut[b] = total;                // out_lens into tail of d_out (int32 bits)
    }
}

// ---------------- phase 0: x (B,T,C) f32 -> xh (B,C,T) f16 ----------------
__global__ void __launch_bounds__(256)
gu_transpose_kernel(const float* __restrict__ x, _Float16* __restrict__ xh) {
    __shared__ float tile[32][33];         // tile[jLocal][chLocal]
    const int b   = blockIdx.z;
    const int j0  = blockIdx.y * 32;
    const int ch0 = blockIdx.x * 32;
    const int tx  = threadIdx.x;           // 32
    const int ty  = threadIdx.y;           // 8
#pragma unroll
    for (int r = 0; r < 32; r += 8)        // coalesced read along channels
        tile[ty + r][tx] = x[((size_t)b * T_TEXT + j0 + ty + r) * CCH + ch0 + tx];
    __syncthreads();

    // packed b32 stores: 2 f16 along j per store
    const int tid = ty * 32 + tx;          // 0..255
    const int jp  = (tid & 15) * 2;        // j pair base 0..30
    const int chh = tid >> 4;              // 0..15
#pragma unroll
    for (int it = 0; it < 2; ++it) {
        const int chl = chh + it * 16;
        union { _Float16 h[2]; unsigned u; } pk;
        pk.h[0] = (_Float16)tile[jp + 0][chl];
        pk.h[1] = (_Float16)tile[jp + 1][chl];
        *(unsigned*)(xh + ((size_t)b * CCH + ch0 + chl) * T_TEXT + j0 + jp) = pk.u;
    }
}

// ---------------- phase 1.5: exact softmax stats (max, 1/sum) ----------------
// Monotone centers -> value-window binary search; dropped terms are exactly 0 in f32.
__global__ void __launch_bounds__(256)
gu_stats_kernel(const float* __restrict__ c, const int* __restrict__ lens,
                float* __restrict__ mArr, float* __restrict__ isArr) {
    const int b = blockIdx.x >> 4;                    // 16 blocks per batch
    const int f = ((blockIdx.x & 15) << 8) + threadIdx.x;
    const int L = lens[b];
    const float t = (f < L) ? (float)f : 0.0f;
    const float* cb = c + b * T_TEXT;                 // scalar base pointer

    // lower bound: first j with cb[j] >= t - WIN
    int lo, hi;
    {
        int l = 0, r = T_TEXT;
        while (l < r) { int mid = (l + r) >> 1; if (cb[mid] < t - WIN) l = mid + 1; else r = mid; }
        lo = l;
    }
    // upper bound: first j with cb[j] > t + WIN
    {
        int l = lo, r = T_TEXT;
        while (l < r) { int mid = (l + r) >> 1; if (cb[mid] <= t + WIN) l = mid + 1; else r = mid; }
        hi = l;
    }

    float best = 3.4e38f;                  // min squared distance within window
    for (int j = lo; j < hi; ++j) {
        float d = t - cb[j];
        best = fminf(best, d * d);
    }
    const float m = -DELTA * best;         // row max of energy
    float s = 0.0f;
    for (int j = lo; j < hi; ++j) {
        float d = t - cb[j];
        s += __expf(-DELTA * d * d - m);
    }
    mArr[b * T_FEAT + f]  = m;
    isArr[b * T_FEAT + f] = 1.0f / s;
}

// ---------------- phase 2: banded attention matmul via WMMA ----------------
// Block = 16 waves; tile = 32 frames x 128 channels; wave = 16x16 via f16 WMMA.
__global__ void __launch_bounds__(512)
gu_attn_kernel(const float* __restrict__ c, const int* __restrict__ lens,
               const float* __restrict__ mArr, const float* __restrict__ isArr,
               const _Float16* __restrict__ xh, float* __restrict__ y) {
    const int b   = blockIdx.z;
    const int f0  = blockIdx.y * 32;
    const int n0  = blockIdx.x * 128;
    const int tid = threadIdx.x;
    const int lane = tid & 31;
    const int w   = tid >> 5;              // 0..15
    const int wf  = w & 1;                 // frame sub-tile (0..1)
    const int wn  = w >> 1;                // channel sub-tile (0..7)

    __shared__ __align__(16) _Float16 pT[32 * P_STRIDE];

    const int L = lens[b];

    // frame-tile t-range for the band test (padded frames use t=0)
    const float tmax = (f0 >= L) ? 0.0f
                                 : (float)((f0 + 31 < L) ? (f0 + 31) : (L - 1));
    const float tmin = (f0 + 31 >= L) ? 0.0f : (float)f0;

    // p-tile producer mapping: each thread -> row fl, 2 adjacent K (one b32 store)
    const int fl = tid >> 4;               // 0..31
    const int kk = (tid & 15) * 2;         // 0,2,...,30
    const int fg = f0 + fl;
    const float t_f  = (fg < L) ? (float)fg : 0.0f;
    const float m_f  = mArr[b * T_FEAT + fg];
    const float is_f = isArr[b * T_FEAT + fg];
    unsigned* pStore = (unsigned*)(pT + fl * P_STRIDE + kk);

    v8f acc = {0.0f, 0.0f, 0.0f, 0.0f, 0.0f, 0.0f, 0.0f, 0.0f};
    const float* cb = c + b * T_TEXT;

    // A fragment: lane -> row M = lane%16; low lanes K{0..7,16..23}, high K{8..15,24..31}
    const int mrow = lane & 15;
    const _Float16* aPtr = pT + (wf * 16 + mrow) * P_STRIDE + ((lane < 16) ? 0 : 8);
    // B fragment: lane -> col N; low lanes K{0..7,16..23}, high K{8..15,24..31}
    const int nCol = n0 + wn * 16 + mrow;
    const _Float16* bBase = xh + ((size_t)b * CCH + nCol) * T_TEXT + ((lane < 16) ? 0 : 8);

    for (int k0 = 0; k0 < T_TEXT; k0 += K_CHUNK) {
        // block-uniform band skip: centers are monotone non-decreasing
        const float cLo = cb[k0];
        const float cHi = cb[k0 + K_CHUNK - 1];
        if (cHi < tmin - RADIUS || cLo > tmax + RADIUS) continue;

        if (k0 + K_CHUNK < T_TEXT)
            __builtin_prefetch(bBase + k0 + K_CHUNK, 0, 1);   // global_prefetch_b8

        // issue B loads early: overlap with transcendental p computation
        union { v16h v; v8h h[2]; } ub;
        ub.h[0] = *(const v8h*)(bBase + k0);
        ub.h[1] = *(const v8h*)(bBase + k0 + 16);

        // build 32x32 p tile in LDS: 2 exps + one packed b32 store per thread
        {
            const float2 cv = *(const float2*)(cb + k0 + kk);
            const float d0 = t_f - cv.x;
            const float d1 = t_f - cv.y;
            union { _Float16 h[2]; unsigned u; } pk;
            pk.h[0] = (_Float16)(__expf(-DELTA * d0 * d0 - m_f) * is_f);
            pk.h[1] = (_Float16)(__expf(-DELTA * d1 * d1 - m_f) * is_f);
            *pStore = pk.u;
        }
        __syncthreads();

        union { v16h v; v8h h[2]; } ua;
        ua.h[0] = *(const v8h*)(aPtr);
        ua.h[1] = *(const v8h*)(aPtr + 16);

        acc = __builtin_amdgcn_wmma_f32_16x16x32_f16(
                  false, ua.v, false, ub.v, (short)0, acc, false, false);
        __syncthreads();                   // protect pT before next overwrite
    }

    // D layout: VGPR r -> M = r (lanes 0-15) / r+8 (lanes 16-31), N = lane%16
    const int fBase = f0 + wf * 16 + ((lane < 16) ? 0 : 8);
    float* yBase = y + ((size_t)b * T_FEAT + fBase) * CCH + nCol;
#pragma unroll
    for (int r = 0; r < 8; ++r)
        yBase[(size_t)r * CCH] = acc[r];
}

// ---------------- launcher ----------------
extern "C" void kernel_launch(void* const* d_in, const int* in_sizes, int n_in,
                              void* d_out, int out_size, void* d_ws, size_t ws_size,
                              hipStream_t stream) {
    const float* x   = (const float*)d_in[0];
    const int*   dur = (const int*)d_in[1];
    // d_in[2] = text_mask (all true), d_in[3] = t_feat scalar: unused

    char* ws = (char*)d_ws;
    const size_t OFF_C    = 0;                                   // 64 KB
    const size_t OFF_LENS = OFF_C    + (size_t)BATCH * T_TEXT * 4;
    const size_t OFF_M    = OFF_LENS + 256;                      // pad to 256B
    const size_t OFF_IS   = OFF_M    + (size_t)BATCH * T_FEAT * 4;
    const size_t OFF_XH   = OFF_IS   + (size_t)BATCH * T_FEAT * 4;

    float*     c     = (float*)(ws + OFF_C);
    int*       lens  = (int*)(ws + OFF_LENS);
    float*     mArr  = (float*)(ws + OFF_M);
    float*     isArr = (float*)(ws + OFF_IS);
    _Float16*  xh    = (_Float16*)(ws + OFF_XH);                 // 16 MB

    float* yOut    = (float*)d_out;
    int*   lensOut = (int*)d_out + Y_ELEMS;

    gu_scan_kernel<<<BATCH, 256, 0, stream>>>(dur, c, lens, lensOut);
    gu_transpose_kernel<<<dim3(CCH / 32, T_TEXT / 32, BATCH), dim3(32, 8), 0, stream>>>(x, xh);
    gu_stats_kernel<<<(BATCH * T_FEAT) / 256, 256, 0, stream>>>(c, lens, mArr, isArr);
    gu_attn_kernel<<<dim3(CCH / 128, T_FEAT / 32, BATCH), 512, 0, stream>>>(
        c, lens, mArr, isArr, xh, yOut);
}
